// RelativeMultiAttention_66408784331311
// MI455X (gfx1250) — matrix operand order repaired and checked by reading, not compiled, not run on hardware
//
#include <hip/hip_runtime.h>

// Transformer-XL style relative multi-head attention for gfx1250 (MI455X).
// Pipeline:
//   (1) convert x, pos_embed -> bf16 ; transpose W_q/k/v/o -> bf16 [n][k]
//   (2) VB[h][c] = v_h . P[c]  (f32, 16x2048 table)
//   (3) QKV GEMM (bf16 WMMA, f32 acc, double-buffered) -> Q, Q+u, K (bf16,
//       [b,h,s,d]) and V transposed (f16, [b,h,d,s])
//   (4) fused flash attention: s_ac via WMMA, rel terms q.P / k.P via small
//       WMMAs with rolling LDS windows + Toeplitz gather, online softmax,
//       P@V via f16 WMMA
//   (5) output projection GEMM (bf16 WMMA, double-buffered) -> f32 d_out

namespace {

constexpr int NH = 16;      // heads
constexpr int DH = 64;      // head dim
constexpr int DM = 1024;    // model dim
constexpr int SQ = 1024;    // sequence
constexpr int NB = 2;       // batch
constexpr int MR = NB * SQ; // 2048 rows
constexpr int PL = 2 * SQ;  // 2048 positions

typedef __attribute__((ext_vector_type(16))) __bf16   v16bf;
typedef __attribute__((ext_vector_type(16))) _Float16 v16h;
typedef __attribute__((ext_vector_type(8)))  float    v8f;

// ---------------- WMMA fragment loaders (wave32; contiguous k order, same for A and B) ----
// A 16x32 (MxK): lane m = lane&15, half = lane>>4.
//   elements e -> K = half*16 + e (contiguous 16)
__device__ inline v16bf load_a_bf(const __bf16* rowbase, int ld) {
  int lane = threadIdx.x & 31;
  int m = lane & 15, hf = lane >> 4;
  const __bf16* p = rowbase + (size_t)m * ld;
  v16bf r;
  ((uint4*)&r)[0] = *(const uint4*)(p + hf * 16);
  ((uint4*)&r)[1] = *(const uint4*)(p + hf * 16 + 8);
  return r;
}
__device__ inline v16h load_a_h(const _Float16* rowbase, int ld) {
  int lane = threadIdx.x & 31;
  int m = lane & 15, hf = lane >> 4;
  const _Float16* p = rowbase + (size_t)m * ld;
  v16h r;
  ((uint4*)&r)[0] = *(const uint4*)(p + hf * 16);
  ((uint4*)&r)[1] = *(const uint4*)(p + hf * 16 + 8);
  return r;
}
// B 32x16 (KxN): lane n = lane&15, half = lane>>4; elements e -> K = half*16+e.
// Source array laid out [n][k] row-major so the 16 K-values are contiguous.
__device__ inline v16bf load_bn_bf(const __bf16* nbase, int ld) {
  int lane = threadIdx.x & 31;
  int n = lane & 15, hf = lane >> 4;
  const __bf16* p = nbase + (size_t)n * ld + hf * 16;
  v16bf r;
  ((uint4*)&r)[0] = ((const uint4*)p)[0];
  ((uint4*)&r)[1] = ((const uint4*)p)[1];
  return r;
}
__device__ inline v16h load_bn_h(const _Float16* nbase, int ld) {
  int lane = threadIdx.x & 31;
  int n = lane & 15, hf = lane >> 4;
  const _Float16* p = nbase + (size_t)n * ld + hf * 16;
  v16h r;
  ((uint4*)&r)[0] = ((const uint4*)p)[0];
  ((uint4*)&r)[1] = ((const uint4*)p)[1];
  return r;
}

__device__ inline v8f wmma_bf(v16bf a, v16bf b, v8f c) {
  return __builtin_amdgcn_wmma_f32_16x16x32_bf16(false, a, false, b, (short)0,
                                                 c, false, false);
}
__device__ inline v8f wmma_h(v16h a, v16h b, v8f c) {
  return __builtin_amdgcn_wmma_f32_16x16x32_f16(false, a, false, b, (short)0,
                                                c, false, false);
}

// ---------------------------- prep kernels --------------------------------
__global__ void __launch_bounds__(256)
k_f32_to_bf16(const float* __restrict__ s, __bf16* __restrict__ d, int n) {
  int i = blockIdx.x * 256 + threadIdx.x;
  if (i < n) d[i] = (__bf16)s[i];
}

// dst[n][k] = (bf16) src[k][n], 1024x1024, LDS-tiled for coalescing.
__global__ void __launch_bounds__(256)
k_transpose_bf16(const float* __restrict__ s, __bf16* __restrict__ d) {
  __shared__ __bf16 tile[32][33];
  int bx = blockIdx.x & 31, by = blockIdx.x >> 5;
  int tx = threadIdx.x & 31, ty = threadIdx.x >> 5;  // 32 x 8
#pragma unroll
  for (int i = 0; i < 4; ++i)
    tile[ty + i * 8][tx] =
        (__bf16)s[(size_t)(by * 32 + ty + i * 8) * DM + bx * 32 + tx];
  __syncthreads();
#pragma unroll
  for (int i = 0; i < 4; ++i)
    d[(size_t)(bx * 32 + ty + i * 8) * DM + by * 32 + tx] =
        tile[tx][ty + i * 8];
}

// VB[h][c] = sum_d v[h][d] * P[c][d]   (f32)
__global__ void __launch_bounds__(256)
k_vb(const float* __restrict__ v, const float* __restrict__ pos,
     float* __restrict__ VB) {
  int i = blockIdx.x * 256 + threadIdx.x;  // 16*2048
  if (i >= NH * PL) return;
  int h = i >> 11, c = i & (PL - 1);
  float a = 0.f;
  for (int d = 0; d < DH; ++d) a += v[h * DH + d] * pos[(size_t)c * DH + d];
  VB[i] = a;
}

// ---------------------------- QKV projection ------------------------------
// grid (N/64, M/256, 3), 256 threads (8 waves). Wave w: 32(M) x 64(N) tile,
// double-buffered A/B fragments so WMMAs overlap in-flight loads.
__global__ void __launch_bounds__(256)
k_qkv_gemm(const __bf16* __restrict__ xbf, const __bf16* __restrict__ wt,
           const float* __restrict__ u, __bf16* __restrict__ Q,
           __bf16* __restrict__ QU, __bf16* __restrict__ K,
           _Float16* __restrict__ Vt) {
  int w = threadIdx.x >> 5;
  int lane = threadIdx.x & 31;
  int ln = lane & 15, lh = lane >> 4;
  int z = blockIdx.z;
  int n0 = blockIdx.x * 64;
  int m0 = blockIdx.y * 256 + w * 32;
  const __bf16* wz = wt + (size_t)z * DM * DM;

  v8f acc[2][4] = {};
  v16bf a[2], b[4], an[2], bn[4];

  auto load_frags = [&](v16bf* A, v16bf* B, int k) {
#pragma unroll
    for (int mt = 0; mt < 2; ++mt)
      A[mt] = load_a_bf(xbf + (size_t)(m0 + mt * 16) * DM + k, DM);
#pragma unroll
    for (int t = 0; t < 4; ++t)
      B[t] = load_bn_bf(wz + (size_t)(n0 + t * 16) * DM + k, DM);
  };
  auto mma_all = [&](v16bf* A, v16bf* B) {
#pragma unroll
    for (int mt = 0; mt < 2; ++mt)
#pragma unroll
      for (int t = 0; t < 4; ++t) acc[mt][t] = wmma_bf(A[mt], B[t], acc[mt][t]);
  };

  load_frags(a, b, 0);
  for (int k = 0; k < DM; k += 64) {
    load_frags(an, bn, k + 32);  // issue next-step loads first
    mma_all(a, b);
    if (k + 64 < DM) load_frags(a, b, k + 64);
    mma_all(an, bn);
  }

#pragma unroll
  for (int mt = 0; mt < 2; ++mt)
#pragma unroll
    for (int t = 0; t < 4; ++t) {
      int n = n0 + t * 16 + ln;
      int hh = n >> 6, d = n & 63;
#pragma unroll
      for (int e = 0; e < 8; ++e) {
        int m = m0 + mt * 16 + lh * 8 + e;
        int bb = m >> 10, s = m & 1023;
        float val = acc[mt][t][e];
        size_t qi = ((size_t)(bb * NH + hh) * SQ + s) * DH + d;
        if (z == 0) {
          Q[qi] = (__bf16)val;
          QU[qi] = (__bf16)(val + u[n]);
        } else if (z == 1) {
          K[qi] = (__bf16)val;
        } else {
          Vt[((size_t)(bb * NH + hh) * DH + d) * SQ + s] = (_Float16)val;
        }
      }
    }
}

// ---------------------------- fused attention -----------------------------
// logits(i,j) = (q_i+u).k_j + q_i.P[1024+j-i] + v_h.P[1024+j-i]
//             + k_j.P[1024+i-j], all / 8; softmax over j; O = P @ V.
// grid = B*NH*(SQ/128) blocks, 256 threads; wave w owns 16 queries.
__global__ void __launch_bounds__(256)
k_attn(const __bf16* __restrict__ Q, const __bf16* __restrict__ QU,
       const __bf16* __restrict__ K, const _Float16* __restrict__ Vt,
       const __bf16* __restrict__ P, const float* __restrict__ VB,
       __bf16* __restrict__ O) {
  __shared__ float vb_sh[PL];                    // 8 KB  term_b row for h
  __shared__ float dqring[8][16][32];            // 16 KB rolling q.P window
  __shared__ float ekbuf[8][16][32];             // 16 KB k.P window
  __shared__ __align__(16) _Float16 pbuf[8][16][32];  // 8 KB probs transpose

  int w = threadIdx.x >> 5;
  int lane = threadIdx.x & 31;
  int ln = lane & 15, lh = lane >> 4;

  int qb = blockIdx.x & 7;
  int h = (blockIdx.x >> 3) & 15;
  int b = blockIdx.x >> 7;
  int i0 = qb * 128 + w * 16;

  for (int i = threadIdx.x; i < PL; i += 256) vb_sh[i] = VB[h * PL + i];
  __syncthreads();

  const __bf16* qbase = Q + ((size_t)(b * NH + h) * SQ + i0) * DH;
  const __bf16* qubase = QU + ((size_t)(b * NH + h) * SQ + i0) * DH;
  const __bf16* kbase = K + ((size_t)(b * NH + h) * SQ) * DH;
  const _Float16* vbase = Vt + ((size_t)(b * NH + h) * DH) * SQ;

  v16bf aq0 = load_a_bf(qbase, DH), aq1 = load_a_bf(qbase + 32, DH);
  v16bf au0 = load_a_bf(qubase, DH), au1 = load_a_bf(qubase + 32, DH);

  float rmax[8], rsum[8];
  v8f occ[4] = {};
#pragma unroll
  for (int e = 0; e < 8; ++e) { rmax[e] = -1e30f; rsum[e] = 0.f; }

  // prologue: low half of the first window, q.P rows [1008-i0, 1024-i0)
  {
    int c0 = 1008 - i0;  // = (1024 + 0 - i0) - 16, in [0,1008]
    v8f z = {};
    v8f dq = wmma_bf(aq0, load_bn_bf(P + (size_t)c0 * DH, DH), z);
    dq = wmma_bf(aq1, load_bn_bf(P + (size_t)c0 * DH + 32, DH), dq);
#pragma unroll
    for (int e = 0; e < 8; ++e)
      dqring[w][lh * 8 + e][(c0 + ln) & 31] = dq[e];
  }

  auto half_step = [&](int j0, float sv[8]) {
    int base = 1024 + j0 - i0;  // in [16, 2032]
    {  // new q.P piece: rows [base, base+16) into the ring
      v8f z = {};
      v8f dq = wmma_bf(aq0, load_bn_bf(P + (size_t)base * DH, DH), z);
      dq = wmma_bf(aq1, load_bn_bf(P + (size_t)base * DH + 32, DH), dq);
#pragma unroll
      for (int e = 0; e < 8; ++e)
        dqring[w][lh * 8 + e][(base + ln) & 31] = dq[e];
    }
    const __bf16* kt = kbase + (size_t)j0 * DH;
    // s_ac = (q+u) . k
    v8f z = {};
    v8f sac = wmma_bf(au0, load_bn_bf(kt, DH), z);
    sac = wmma_bf(au1, load_bn_bf(kt + 32, DH), sac);
    // k.P over window [base2-16, base2+16), base2 = 1024 + i0 - j0
    v16bf ak0 = load_a_bf(kt, DH), ak1 = load_a_bf(kt + 32, DH);
    int base2 = 1024 + i0 - j0;
#pragma unroll
    for (int f = 0; f < 2; ++f) {
      int cb = base2 - 16 + f * 16;  // in [0, 2032]
      v8f zz = {};
      v8f ek = wmma_bf(ak0, load_bn_bf(P + (size_t)cb * DH, DH), zz);
      ek = wmma_bf(ak1, load_bn_bf(P + (size_t)cb * DH + 32, DH), ek);
#pragma unroll
      for (int e = 0; e < 8; ++e)
        ekbuf[w][lh * 8 + e][f * 16 + ln] = ek[e];
    }
    // Toeplitz gathers:  c = base + nj - mi ; c2-offset = 16 + mi - nj
#pragma unroll
    for (int e = 0; e < 8; ++e) {
      int mi = lh * 8 + e;
      float t = sac[e];
      t += dqring[w][mi][(base + ln - mi) & 31];
      t += ekbuf[w][ln][16 + mi - ln];
      t += vb_sh[base + ln - mi];
      sv[e] = t * 0.125f;  // / sqrt(64)
    }
  };

  for (int j32 = 0; j32 < SQ; j32 += 32) {
    if (j32 + 32 < SQ) {
      __builtin_prefetch(kbase + (size_t)(j32 + 32) * DH, 0, 1);
      __builtin_prefetch(vbase + (size_t)(ln * 4) * SQ + j32 + 32, 0, 1);
    }
    float sv0[8], sv1[8];
    half_step(j32, sv0);
    half_step(j32 + 16, sv1);
    // online softmax over the 32-key chunk (rows live in 16-lane halves)
#pragma unroll
    for (int e = 0; e < 8; ++e) {
      float mx = fmaxf(sv0[e], sv1[e]);
#pragma unroll
      for (int msk = 1; msk < 16; msk <<= 1)
        mx = fmaxf(mx, __shfl_xor(mx, msk, 32));
      float nm = fmaxf(rmax[e], mx);
      float sc = __expf(rmax[e] - nm);
      rmax[e] = nm;
      float p0 = __expf(sv0[e] - nm);
      float p1 = __expf(sv1[e] - nm);
      float rs = p0 + p1;
#pragma unroll
      for (int msk = 1; msk < 16; msk <<= 1) rs += __shfl_xor(rs, msk, 32);
      rsum[e] = rsum[e] * sc + rs;
      pbuf[w][lh * 8 + e][ln] = (_Float16)p0;
      pbuf[w][lh * 8 + e][16 + ln] = (_Float16)p1;
#pragma unroll
      for (int t = 0; t < 4; ++t) occ[t][e] = occ[t][e] * sc;
    }
    // P @ V : probs A-frag (16x32 f16) x V B-frags (32x16 f16, d-major)
    v16h ap = load_a_h(&pbuf[w][0][0], 32);
#pragma unroll
    for (int t = 0; t < 4; ++t) {
      v16h bv = load_bn_h(vbase + (size_t)(t * 16) * SQ + j32, SQ);
      occ[t] = wmma_h(ap, bv, occ[t]);
    }
  }

  size_t obase = ((size_t)b * SQ + i0) * DM + h * DH;
#pragma unroll
  for (int t = 0; t < 4; ++t)
#pragma unroll
    for (int e = 0; e < 8; ++e)
      O[obase + (size_t)(lh * 8 + e) * DM + t * 16 + ln] =
          (__bf16)(occ[t][e] / rsum[e]);
}

// ---------------------------- output projection ---------------------------
// Same double-buffered scheme as k_qkv_gemm; writes f32 directly to d_out.
__global__ void __launch_bounds__(256)
k_out_gemm(const __bf16* __restrict__ A, const __bf16* __restrict__ wt,
           float* __restrict__ out) {
  int w = threadIdx.x >> 5;
  int lane = threadIdx.x & 31;
  int ln = lane & 15, lh = lane >> 4;
  int n0 = blockIdx.x * 64;
  int m0 = blockIdx.y * 256 + w * 32;

  v8f acc[2][4] = {};
  v16bf a[2], b[4], an[2], bn[4];

  auto load_frags = [&](v16bf* Af, v16bf* Bf, int k) {
#pragma unroll
    for (int mt = 0; mt < 2; ++mt)
      Af[mt] = load_a_bf(A + (size_t)(m0 + mt * 16) * DM + k, DM);
#pragma unroll
    for (int t = 0; t < 4; ++t)
      Bf[t] = load_bn_bf(wt + (size_t)(n0 + t * 16) * DM + k, DM);
  };
  auto mma_all = [&](v16bf* Af, v16bf* Bf) {
#pragma unroll
    for (int mt = 0; mt < 2; ++mt)
#pragma unroll
      for (int t = 0; t < 4; ++t)
        acc[mt][t] = wmma_bf(Af[mt], Bf[t], acc[mt][t]);
  };

  load_frags(a, b, 0);
  for (int k = 0; k < DM; k += 64) {
    load_frags(an, bn, k + 32);
    mma_all(a, b);
    if (k + 64 < DM) load_frags(a, b, k + 64);
    mma_all(an, bn);
  }

#pragma unroll
  for (int mt = 0; mt < 2; ++mt)
#pragma unroll
    for (int t = 0; t < 4; ++t)
#pragma unroll
      for (int e = 0; e < 8; ++e)
        out[(size_t)(m0 + mt * 16 + lh * 8 + e) * DM + n0 + t * 16 + ln] =
            acc[mt][t][e];
}

}  // namespace

extern "C" void kernel_launch(void* const* d_in, const int* in_sizes, int n_in,
                              void* d_out, int out_size, void* d_ws,
                              size_t ws_size, hipStream_t stream) {
  const float* x = (const float*)d_in[0];
  const float* W_q = (const float*)d_in[1];
  const float* W_k = (const float*)d_in[2];
  const float* W_v = (const float*)d_in[3];
  const float* W_o = (const float*)d_in[4];
  const float* u = (const float*)d_in[5];
  const float* vb = (const float*)d_in[6];
  const float* pos = (const float*)d_in[7];
  float* out = (float*)d_out;

  // workspace carve-up (256B aligned)
  char* p = (char*)d_ws;
  auto take = [&](size_t bytes) {
    char* r = p;
    p += (bytes + 255) & ~(size_t)255;
    return r;
  };
  __bf16* xbf = (__bf16*)take((size_t)MR * DM * 2);          // 4 MB
  __bf16* wtq = (__bf16*)take((size_t)3 * DM * DM * 2);      // 6 MB (q,k,v)
  __bf16* wto = (__bf16*)take((size_t)DM * DM * 2);          // 2 MB
  __bf16* pbf = (__bf16*)take((size_t)PL * DH * 2);          // 256 KB
  __bf16* Q = (__bf16*)take((size_t)NB * NH * SQ * DH * 2);  // 4 MB
  __bf16* QU = (__bf16*)take((size_t)NB * NH * SQ * DH * 2);
  __bf16* K = (__bf16*)take((size_t)NB * NH * SQ * DH * 2);
  _Float16* Vt = (_Float16*)take((size_t)NB * NH * SQ * DH * 2);
  __bf16* O = (__bf16*)take((size_t)MR * DM * 2);            // 4 MB
  float* VB = (float*)take((size_t)NH * PL * 4);             // 128 KB

  // (1) conversions / transposes
  k_f32_to_bf16<<<(MR * DM + 255) / 256, 256, 0, stream>>>(x, xbf, MR * DM);
  k_f32_to_bf16<<<(PL * DH + 255) / 256, 256, 0, stream>>>(pos, pbf, PL * DH);
  k_transpose_bf16<<<1024, 256, 0, stream>>>(W_q, wtq);
  k_transpose_bf16<<<1024, 256, 0, stream>>>(W_k, wtq + (size_t)DM * DM);
  k_transpose_bf16<<<1024, 256, 0, stream>>>(W_v, wtq + (size_t)2 * DM * DM);
  k_transpose_bf16<<<1024, 256, 0, stream>>>(W_o, wto);
  // (2) term_b table
  k_vb<<<(NH * PL) / 256, 256, 0, stream>>>(vb, pos, VB);
  // (3) QKV projections
  k_qkv_gemm<<<dim3(DM / 64, MR / 256, 3), 256, 0, stream>>>(xbf, wtq, u, Q,
                                                             QU, K, Vt);
  // (4) fused relative attention
  k_attn<<<NB * NH * (SQ / 128), 256, 0, stream>>>(Q, QU, K, Vt, pbf, VB, O);
  // (5) output projection
  k_out_gemm<<<dim3(DM / 64, MR / 256), 256, 0, stream>>>(O, wto, out);
}
